// LoFTRWConfSinview_74371653697938
// MI455X (gfx1250) — matrix operand order repaired and checked by reading, not compiled, not run on hardware
//
#include <hip/hip_runtime.h>
#include <hip/hip_bf16.h>

// ---------------- constants ----------------
#define BB   4
#define HH   512
#define WW   512
#define DD   320
#define NHD  8
#define DHD  40
#define HS   64
#define LSEQ 4096   // 64*64

// ---------------- device helpers ----------------
__device__ __forceinline__ unsigned short f2bf(float f) {
  unsigned int u = __float_as_uint(f);
  unsigned int r = (u + 0x7FFFu + ((u >> 16) & 1u)) >> 16;
  return (unsigned short)r;
}
__device__ __forceinline__ float bf2f(unsigned short s) {
  return __uint_as_float(((unsigned int)s) << 16);
}
__device__ __forceinline__ float elu1(float x) {
  return x > 0.f ? x + 1.f : __expf(x);
}

#if defined(__HIP_DEVICE_COMPILE__)
typedef __attribute__((ext_vector_type(16))) __bf16          v16bf;
typedef __attribute__((ext_vector_type(8)))  float           v8f;
typedef __attribute__((ext_vector_type(8)))  unsigned short  us8;
#endif

// ---------------- WMMA GEMM ----------------
// C[M,N] (f32, optional bf16 copy) = A[M,K](bf16) x B (bf16, [K,N] or [N,K] if transB)
// optional bias[N] and ReLU fused in the epilogue. grid.z batches via element strides.
//
// LDS layout chosen so each lane's WMMA fragment is contiguous:
//   As[row][k]  : A-frag = 2x 16B runs (K = kg*16..+16, contiguous)        -> ds_load_b128
//   BsT[n][k]   : B-frag = 2x 16B runs (K = kg*16..+16, column-major tile) -> ds_load_b128
// Staging uses one 16B global load per thread per tile when 8|K; ds_store_b128 (8x b16 if non-transB B).
__global__ __launch_bounds__(256) void k_gemm(
    const unsigned short* __restrict__ A, const unsigned short* __restrict__ Bm,
    float* __restrict__ C, unsigned short* __restrict__ Cb,
    const float* __restrict__ bias, int M, int N, int K,
    int transB, int relu, long long sA, long long sB, long long sC) {
#if defined(__HIP_DEVICE_COMPILE__)
  __shared__ alignas(16) unsigned short As[64][40];   // 80B row stride (16B aligned)
  __shared__ alignas(16) unsigned short BsT[64][40];
  const int z = blockIdx.z;
  A += (long long)z * sA;
  Bm += (long long)z * sB;
  C += (long long)z * sC;
  const int m0 = blockIdx.y * 64;
  const int n0 = blockIdx.x * 64;
  const int tid  = threadIdx.x;
  const int lane = tid & 31;
  const int wave = tid >> 5;
  const int rslab = (wave >> 1) << 4;  // 0,16,32,48
  const int cslab = (wave & 1) << 5;   // 0,32
  const int r16 = lane & 15;
  const int kg  = lane >> 4;
  const bool vec = ((K & 7) == 0);
  v8f acc0 = {};
  v8f acc1 = {};
  for (int k0 = 0; k0 < K; k0 += 32) {
    if (vec) {
      // ---- A tile 64x32: one 16B chunk per thread ----
      {
        int r = tid >> 2;            // 0..63
        int koff = (tid & 3) << 3;   // 0,8,16,24
        us8 av = {};
        if (m0 + r < M && k0 + koff + 8 <= K)
          av = *(const us8*)&A[(long long)(m0 + r) * K + (k0 + koff)];
        *(us8*)&As[r][koff] = av;
      }
      // ---- B tile (stored transposed BsT[n][k]) ----
      if (transB) {
        int n = tid >> 2;            // 0..63
        int kf = (tid & 3) << 3;     // 0,8,16,24
        us8 bv = {};
        if (n0 + n < N && k0 + kf + 8 <= K)
          bv = *(const us8*)&Bm[(long long)(n0 + n) * K + (k0 + kf)];
        *(us8*)&BsT[n][kf] = bv;
      } else {
        int k = tid >> 3;            // 0..31
        int noff = (tid & 7) << 3;   // 0..56
        us8 bv = {};
        if (k0 + k < K) {
          if (n0 + noff + 8 <= N) {
            bv = *(const us8*)&Bm[(long long)(k0 + k) * N + (n0 + noff)];
          } else {
            for (int e = 0; e < 8; ++e)
              bv[e] = (n0 + noff + e < N) ? Bm[(long long)(k0 + k) * N + (n0 + noff + e)]
                                          : (unsigned short)0;
          }
        }
        // transpose scatter into LDS (8x ds_store_b16)
        #pragma unroll
        for (int e = 0; e < 8; ++e) BsT[noff + e][k] = bv[e];
        // mark chunk consumed; nothing else to do
        (void)bv;
      }
    } else {
      // ---- scalar fallback (only conv1: K=27) ----
      for (int t = tid; t < 64 * 32; t += 256) {
        int r = t >> 5, k = t & 31;
        unsigned short v = 0;
        if (m0 + r < M && k0 + k < K) v = A[(long long)(m0 + r) * K + (k0 + k)];
        As[r][k] = v;
      }
      for (int t = tid; t < 64 * 32; t += 256) {
        int n = t >> 5, k = t & 31;
        unsigned short v = 0;
        if (k0 + k < K && n0 + n < N)
          v = transB ? Bm[(long long)(n0 + n) * K + (k0 + k)]
                     : Bm[(long long)(k0 + k) * N + (n0 + n)];
        BsT[n][k] = v;
      }
    }
    if (k0 + 32 < K && m0 + (tid & 63) < M)
      __builtin_prefetch(&A[(long long)(m0 + (tid & 63)) * K + (k0 + 32)], 0, 1);
    __syncthreads();
    // ---- fragments: contiguous 16B LDS reads ----
    union FA { v16bf v; us8 h[2]; } ua;
    ua.h[0] = *(const us8*)&As[rslab + r16][kg * 16];
    ua.h[1] = *(const us8*)&As[rslab + r16][kg * 16 + 8];
    union FB { v16bf v; us8 h[2]; } ub0, ub1;
    ub0.h[0] = *(const us8*)&BsT[cslab + r16][kg * 16];
    ub0.h[1] = *(const us8*)&BsT[cslab + r16][kg * 16 + 8];
    ub1.h[0] = *(const us8*)&BsT[cslab + 16 + r16][kg * 16];
    ub1.h[1] = *(const us8*)&BsT[cslab + 16 + r16][kg * 16 + 8];
    acc0 = __builtin_amdgcn_wmma_f32_16x16x32_bf16(false, ua.v, false, ub0.v,
                                                   (short)0, acc0, false, false);
    acc1 = __builtin_amdgcn_wmma_f32_16x16x32_bf16(false, ua.v, false, ub1.v,
                                                   (short)0, acc1, false, false);
    __syncthreads();
  }
  // epilogue: bias / relu / f32 store / optional bf16 store
  unsigned short* Cbz = Cb ? Cb + (long long)z * sC : (unsigned short*)0;
  #pragma unroll
  for (int v = 0; v < 8; ++v) {
    int m = m0 + rslab + v + ((lane >> 4) << 3);
    if (m >= M) continue;
    int na = n0 + cslab + (lane & 15);
    int nb = na + 16;
    if (na < N) {
      float val = acc0[v];
      if (bias) val += bias[na];
      if (relu) val = fmaxf(val, 0.f);
      C[(long long)m * N + na] = val;
      if (Cbz) Cbz[(long long)m * N + na] = f2bf(val);
    }
    if (nb < N) {
      float val = acc1[v];
      if (bias) val += bias[nb];
      if (relu) val = fmaxf(val, 0.f);
      C[(long long)m * N + nb] = val;
      if (Cbz) Cbz[(long long)m * N + nb] = f2bf(val);
    }
  }
#endif
}

// ---------------- elementwise / staging kernels ----------------
__global__ void k_zero(float* p, int n) {
  int i = blockIdx.x * blockDim.x + threadIdx.x;
  if (i < n) p[i] = 0.f;
}

__global__ void k_f32_to_bf16(const float* __restrict__ in, unsigned short* __restrict__ out,
                              long long n) {
  long long i = (long long)blockIdx.x * blockDim.x + threadIdx.x;
  if (i < n) out[i] = f2bf(in[i]);
}

__global__ void k_mask_fill(const float* __restrict__ img, const int* __restrict__ mask,
                            const float* __restrict__ token, float* __restrict__ x0,
                            long long total) {
  long long idx = (long long)blockIdx.x * blockDim.x + threadIdx.x;
  if (idx >= total) return;
  int x = (int)(idx % WW);
  long long t = idx / WW;
  int y = (int)(t % HH); t /= HH;
  int c = (int)(t % 3);
  int n = (int)(t / 3);
  float m = (float)mask[((long long)n * 16 + (y >> 5)) * 16 + (x >> 5)];
  x0[idx] = img[idx] * (1.f - m) + token[c] * m;
}

// im2col for 3x3 stride-2 SAME(pad 0,1) conv; writes bf16 patches [B*OH*OW, C*9]
__global__ void k_im2col(const float* __restrict__ in, unsigned short* __restrict__ col,
                         int Bn, int Cc, int H, int W, int OH, int OW, int nchw) {
  long long idx = (long long)blockIdx.x * blockDim.x + threadIdx.x;
  long long total = (long long)Bn * OH * OW * Cc * 9;
  if (idx >= total) return;
  int kc = (int)(idx % (Cc * 9));
  long long row = idx / (Cc * 9);
  int ox = (int)(row % OW);
  long long t = row / OW;
  int oy = (int)(t % OH);
  int n  = (int)(t / OH);
  int c  = kc / 9;
  int kk = kc % 9;
  int ky = kk / 3, kx = kk % 3;
  int iy = oy * 2 + ky, ix = ox * 2 + kx;
  float v = 0.f;
  if (iy < H && ix < W)
    v = nchw ? in[(((long long)n * Cc + c) * H + iy) * W + ix]
             : in[(((long long)n * H + iy) * W + ix) * Cc + c];
  col[idx] = f2bf(v);
}

// add sinusoidal pos-encoding to seq-layout features, emit f32 + bf16
__global__ void k_posenc(const float* __restrict__ pre, float* __restrict__ seq,
                         unsigned short* __restrict__ seqb) {
  long long idx = (long long)blockIdx.x * blockDim.x + threadIdx.x;
  long long total = (long long)BB * LSEQ * DD;
  if (idx >= total) return;
  int ch = (int)(idx % DD);
  long long t = idx / DD;
  int l = (int)(t % LSEQ);
  int y = l >> 6, x = l & 63;
  int g = ch >> 2, r = ch & 3;
  const float LC = -0.05756462732485115f;  // -ln(10000)/160
  float dv = __expf((float)g * 2.f * LC);
  float arg = ((r < 2) ? (float)(x + 1) : (float)(y + 1)) * dv;
  float pe = (r & 1) ? __cosf(arg) : __sinf(arg);
  float v = pre[idx] + pe;
  seq[idx] = v;
  seqb[idx] = f2bf(v);
}

// row layernorm over C, optional residual add, optional bf16 mirror
__global__ __launch_bounds__(256) void k_layernorm(
    const float* __restrict__ in, float* __restrict__ out, unsigned short* __restrict__ outb,
    const float* __restrict__ gamma, const float* __restrict__ beta,
    const float* __restrict__ resid, int C) {
  __shared__ float s1[256], s2[256];
  long long row = blockIdx.x;
  const float* x = in + row * C;
  float a = 0.f, b = 0.f;
  for (int c = threadIdx.x; c < C; c += 256) { float v = x[c]; a += v; b += v * v; }
  s1[threadIdx.x] = a; s2[threadIdx.x] = b;
  __syncthreads();
  for (int st = 128; st > 0; st >>= 1) {
    if (threadIdx.x < st) { s1[threadIdx.x] += s1[threadIdx.x + st]; s2[threadIdx.x] += s2[threadIdx.x + st]; }
    __syncthreads();
  }
  float mu = s1[0] / C;
  float var = s2[0] / C - mu * mu;
  float rs = rsqrtf(var + 1e-5f);
  for (int c = threadIdx.x; c < C; c += 256) {
    float v = (x[c] - mu) * rs * gamma[c] + beta[c];
    if (resid) v += resid[row * C + c];
    out[row * C + c] = v;
    if (outb) outb[row * C + c] = f2bf(v);
  }
}

// linear attention pass A: kv[n,h,d,v], ksum[n,h,d] (one block per (n,h))
__global__ __launch_bounds__(256) void k_la_kv(
    const float* __restrict__ kin, const float* __restrict__ vin,
    float* __restrict__ kv, float* __restrict__ ksum, int S) {
  __shared__ float sk[DHD], sv[DHD];
  int n = blockIdx.x / NHD, hh = blockIdx.x % NHD;
  int tid = threadIdx.x;
  float acc[7];
  #pragma unroll
  for (int j = 0; j < 7; ++j) acc[j] = 0.f;
  float ks = 0.f;
  for (int s = 0; s < S; ++s) {
    if (tid < DHD) sk[tid] = elu1(kin[((long long)n * S + s) * DD + hh * DHD + tid]);
    else if (tid >= 64 && tid < 64 + DHD) sv[tid - 64] = vin[((long long)n * S + s) * DD + hh * DHD + (tid - 64)];
    __syncthreads();
    int j = 0;
    for (int p = tid; p < DHD * DHD; p += 256, ++j) acc[j] += sk[p / DHD] * sv[p % DHD];
    if (tid < DHD) ks += sk[tid];
    __syncthreads();
  }
  int j = 0;
  for (int p = tid; p < DHD * DHD; p += 256, ++j)
    kv[(((long long)n * NHD + hh) * DHD + p / DHD) * DHD + (p % DHD)] = acc[j];
  if (tid < DHD) ksum[((long long)n * NHD + hh) * DHD + tid] = ks;
}

// linear attention pass B: msg[n,l,:] (one block of 320 per (n,l))
__global__ __launch_bounds__(320) void k_la_msg(
    const float* __restrict__ qin, const float* __restrict__ kv,
    const float* __restrict__ ksum, float* __restrict__ msg, int Lq) {
  __shared__ float qe[DD];
  __shared__ float zs[NHD];
  long long nl = blockIdx.x;
  int n = (int)(nl / Lq);
  int tid = threadIdx.x;
  qe[tid] = elu1(qin[nl * DD + tid]);
  __syncthreads();
  int hh = tid / DHD, dv = tid % DHD;
  if (dv == 0) {
    float den = 0.f;
    const float* kss = ksum + ((long long)n * NHD + hh) * DHD;
    for (int dh = 0; dh < DHD; ++dh) den += qe[hh * DHD + dh] * kss[dh];
    zs[hh] = 1.f / (den + 1e-6f);
  }
  __syncthreads();
  float sacc = 0.f;
  const float* kvh = kv + ((long long)n * NHD + hh) * DHD * DHD;
  for (int dh = 0; dh < DHD; ++dh) sacc += qe[hh * DHD + dh] * kvh[dh * DHD + dv];
  msg[nl * DD + tid] = sacc * zs[hh];
}

__global__ void k_concat_bf16(const float* __restrict__ a, const float* __restrict__ b,
                              unsigned short* __restrict__ out, long long rows, int Ca, int Cb) {
  long long idx = (long long)blockIdx.x * blockDim.x + threadIdx.x;
  long long total = rows * (Ca + Cb);
  if (idx >= total) return;
  int c = (int)(idx % (Ca + Cb));
  long long r = idx / (Ca + Cb);
  float v = (c < Ca) ? a[r * Ca + c] : b[r * Cb + (c - Ca)];
  out[idx] = f2bf(v);
}

__global__ __launch_bounds__(256) void k_softmax_bf16(
    const float* __restrict__ in, unsigned short* __restrict__ out, int C, float scale) {
  __shared__ float red[256];
  long long row = blockIdx.x;
  const float* x = in + row * (long long)C;
  float mx = -3.4e38f;
  for (int c = threadIdx.x; c < C; c += 256) mx = fmaxf(mx, x[c] * scale);
  red[threadIdx.x] = mx;
  __syncthreads();
  for (int st = 128; st > 0; st >>= 1) {
    if (threadIdx.x < st) red[threadIdx.x] = fmaxf(red[threadIdx.x], red[threadIdx.x + st]);
    __syncthreads();
  }
  mx = red[0];
  __syncthreads();
  float sum = 0.f;
  for (int c = threadIdx.x; c < C; c += 256) sum += __expf(x[c] * scale - mx);
  red[threadIdx.x] = sum;
  __syncthreads();
  for (int st = 128; st > 0; st >>= 1) {
    if (threadIdx.x < st) red[threadIdx.x] += red[threadIdx.x + st];
    __syncthreads();
  }
  float inv = 1.f / red[0];
  for (int c = threadIdx.x; c < C; c += 256)
    out[row * (long long)C + c] = f2bf(__expf(x[c] * scale - mx) * inv);
}

__global__ void k_pixshuf_rgb(const float* __restrict__ oseq, float* __restrict__ rgb) {
  long long idx = (long long)blockIdx.x * blockDim.x + threadIdx.x;
  long long total = (long long)BB * 3 * HH * WW;
  if (idx >= total) return;
  int X = (int)(idx % WW);
  long long t = idx / WW;
  int Y = (int)(t % HH); t /= HH;
  int oc = (int)(t % 3);
  int n = (int)(t / 3);
  int y = Y >> 3, ry = Y & 7, x = X >> 3, rx = X & 7;
  int ch = oc * 64 + ry * 8 + rx;
  rgb[idx] = oseq[((long long)n * LSEQ + y * HS + x) * 192 + ch];
}

__global__ void k_refine(const float* __restrict__ rseq, const float* __restrict__ rgb,
                         float* __restrict__ refined, float* __restrict__ rel) {
  long long idx = (long long)blockIdx.x * blockDim.x + threadIdx.x;
  long long total = (long long)BB * HH * WW;
  if (idx >= total) return;
  int X = (int)(idx % WW);
  long long t = idx / WW;
  int Y = (int)(t % HH);
  int n = (int)(t / HH);
  int y = Y >> 3, ry = Y & 7, x = X >> 3, rx = X & 7;
  long long base = ((long long)n * LSEQ + y * HS + x) * 256;
  float r0 = rseq[base + ry * 8 + rx];
  float rl = 1.f / (1.f + __expf(-r0));
  rel[idx] = rl;
  for (int c = 0; c < 3; ++c) {
    float dlt = rseq[base + (1 + c) * 64 + ry * 8 + rx];
    long long pi = (((long long)n * 3 + c) * HH + Y) * WW + X;
    refined[pi] = rgb[pi] * (1.f - rl) + dlt * rl;
  }
}

__global__ __launch_bounds__(256) void k_loss_reduce(
    const float* __restrict__ img, const float* __restrict__ rgb,
    const float* __restrict__ refined, const int* __restrict__ mask,
    const float* __restrict__ msup, float* __restrict__ accum) {
  __shared__ float red[256];
  long long idx = (long long)blockIdx.x * 256 + threadIdx.x;
  long long total = (long long)BB * HH * WW;
  float vals[5] = {0.f, 0.f, 0.f, 0.f, 0.f};
  if (idx < total) {
    int X = (int)(idx % WW);
    long long t = idx / WW;
    int Y = (int)(t % HH);
    int n = (int)(t / HH);
    float mf = (float)mask[((long long)n * 16 + (Y >> 5)) * 16 + (X >> 5)];
    float mp = mf * msup[idx];
    float l1 = 0.f, l1r = 0.f;
    for (int c = 0; c < 3; ++c) {
      long long pi = (((long long)n * 3 + c) * HH + Y) * WW + X;
      l1 += fabsf(img[pi] - rgb[pi]);
      l1r += fabsf(img[pi] - refined[pi]);
    }
    vals[0] = l1 * mp; vals[1] = mp; vals[2] = l1 * (1.f - mf);
    vals[3] = 1.f - mf; vals[4] = l1r * mp;
  }
  for (int i = 0; i < 5; ++i) {
    red[threadIdx.x] = vals[i];
    __syncthreads();
    for (int st = 128; st > 0; st >>= 1) {
      if (threadIdx.x < st) red[threadIdx.x] += red[threadIdx.x + st];
      __syncthreads();
    }
    if (threadIdx.x == 0) atomicAdd(&accum[i], red[0]);
    __syncthreads();
  }
}

__global__ void k_finalize(const float* __restrict__ accum, float* __restrict__ loss) {
  if (blockIdx.x == 0 && threadIdx.x == 0) {
    float l = accum[0] / (accum[1] + 1e-5f) / 3.f;
    l += 0.05f * accum[2] / (accum[3] + 1e-5f) / 3.f;
    l += 0.25f * accum[4] / (accum[1] + 1e-5f) / 3.f;
    loss[0] = l;
  }
}

// ---------------- host-side helpers ----------------
static inline unsigned gdiv(long long n, int b) { return (unsigned)((n + b - 1) / b); }

static inline void launch_gemm(hipStream_t st, const void* A, const void* Bm, void* C, void* Cb,
                               const void* bias, int M, int N, int K, int transB, int relu,
                               long long sA, long long sB, long long sC, int batch) {
  dim3 g((unsigned)((N + 63) / 64), (unsigned)((M + 63) / 64), (unsigned)batch);
  k_gemm<<<g, 256, 0, st>>>((const unsigned short*)A, (const unsigned short*)Bm,
                            (float*)C, (unsigned short*)Cb, (const float*)bias,
                            M, N, K, transB, relu, sA, sB, sC);
}
static inline void launch_cvt(hipStream_t st, const void* src, void* dst, long long n) {
  k_f32_to_bf16<<<gdiv(n, 256), 256, 0, st>>>((const float*)src, (unsigned short*)dst, n);
}

struct EncTmp {
  float *q, *k, *v, *msg, *ffn1, *ffn2;
  unsigned short *msgb, *cat1, *ffn1b;
  float *kvbuf, *ksumbuf;
};

static void run_encoder(hipStream_t st, float* xf, unsigned short* xb,
                        const unsigned short* srcb,
                        const unsigned short* wq, const unsigned short* wk,
                        const unsigned short* wv, const unsigned short* wm,
                        const unsigned short* w1, const unsigned short* w2,
                        const float* g1, const float* b1, const float* g2, const float* b2,
                        const EncTmp& t) {
  const int M = BB * LSEQ;
  launch_gemm(st, xb,   wq, t.q, nullptr, nullptr, M, DD, DD, 0, 0, 0, 0, 0, 1);
  launch_gemm(st, srcb, wk, t.k, nullptr, nullptr, M, DD, DD, 0, 0, 0, 0, 0, 1);
  launch_gemm(st, srcb, wv, t.v, nullptr, nullptr, M, DD, DD, 0, 0, 0, 0, 0, 1);
  k_la_kv<<<BB * NHD, 256, 0, st>>>(t.k, t.v, t.kvbuf, t.ksumbuf, LSEQ);
  k_la_msg<<<M, DD, 0, st>>>(t.q, t.kvbuf, t.ksumbuf, t.msg, LSEQ);
  launch_cvt(st, t.msg, t.msgb, (long long)M * DD);
  launch_gemm(st, t.msgb, wm, t.ffn2 /* mm scratch */, nullptr, nullptr, M, DD, DD, 0, 0, 0, 0, 0, 1);
  k_layernorm<<<M, 256, 0, st>>>(t.ffn2, t.msg, nullptr, g1, b1, nullptr, DD);
  k_concat_bf16<<<gdiv((long long)M * 2 * DD, 256), 256, 0, st>>>(xf, t.msg, t.cat1, M, DD, DD);
  launch_gemm(st, t.cat1, w1, t.ffn1, t.ffn1b, nullptr, M, 2 * DD, 2 * DD, 0, 1, 0, 0, 0, 1);
  launch_gemm(st, t.ffn1b, w2, t.ffn2, nullptr, nullptr, M, DD, 2 * DD, 0, 0, 0, 0, 0, 1);
  k_layernorm<<<M, 256, 0, st>>>(t.ffn2, xf, xb, g2, b2, xf, DD);  // x + msg, in place
}

// ---------------- entry ----------------
extern "C" void kernel_launch(void* const* d_in, const int* in_sizes, int n_in,
                              void* d_out, int out_size, void* d_ws, size_t ws_size,
                              hipStream_t stream) {
  (void)in_sizes; (void)n_in; (void)out_size; (void)ws_size;
  const float* img      = (const float*)d_in[0];
  const int*   mask     = (const int*)d_in[1];
  const float* img2     = (const float*)d_in[2];
  const float* masksup  = (const float*)d_in[3];
  const float* mtoken   = (const float*)d_in[4];
  const float* bb_w1    = (const float*)d_in[5];
  const float* bb_b1    = (const float*)d_in[6];
  const float* bb_w2    = (const float*)d_in[7];
  const float* bb_b2    = (const float*)d_in[8];
  const float* bb_w3    = (const float*)d_in[9];
  const float* bb_b3    = (const float*)d_in[10];
  const float* Wq       = (const float*)d_in[11];
  const float* Wk       = (const float*)d_in[12];
  const float* Wv       = (const float*)d_in[13];
  const float* Wm       = (const float*)d_in[14];
  const float* W1       = (const float*)d_in[15];
  const float* W2       = (const float*)d_in[16];
  const float* g1       = (const float*)d_in[17];
  const float* b1       = (const float*)d_in[18];
  const float* g2       = (const float*)d_in[19];
  const float* b2       = (const float*)d_in[20];
  const float* out_w    = (const float*)d_in[21];
  const float* out_b    = (const float*)d_in[22];
  const float* ref_w    = (const float*)d_in[23];
  const float* ref_b    = (const float*)d_in[24];
  float* out = (float*)d_out;

  char* ws = (char*)d_ws;
  size_t off = 0;
  auto alloc = [&](size_t bytes) -> char* {
    size_t p = off;
    off = (off + bytes + 255) & ~(size_t)255;
    return ws + p;
  };

  const int M1 = BB * 256 * 256, K1 = 27,   N1 = 128;
  const int M2 = BB * 128 * 128, K2 = 1152, N2 = 256;
  const int M3 = BB * 64 * 64,   K3 = 2304, N3 = DD;
  const long long SEQE = (long long)BB * LSEQ * DD;       // 5,242,880
  const long long SIME = (long long)BB * LSEQ * LSEQ;     // 67,108,864

  // bf16 weights
  unsigned short* bw1   = (unsigned short*)alloc((size_t)N1 * K1 * 2);
  unsigned short* bw2   = (unsigned short*)alloc((size_t)N2 * K2 * 2);
  unsigned short* bw3   = (unsigned short*)alloc((size_t)N3 * K3 * 2);
  unsigned short* wqb   = (unsigned short*)alloc((size_t)4 * DD * DD * 2);
  unsigned short* wkb   = (unsigned short*)alloc((size_t)4 * DD * DD * 2);
  unsigned short* wvb   = (unsigned short*)alloc((size_t)4 * DD * DD * 2);
  unsigned short* wmb   = (unsigned short*)alloc((size_t)4 * DD * DD * 2);
  unsigned short* w1b   = (unsigned short*)alloc((size_t)4 * 640 * 640 * 2);
  unsigned short* w2b   = (unsigned short*)alloc((size_t)4 * 640 * 320 * 2);
  unsigned short* outwb = (unsigned short*)alloc((size_t)192 * 320 * 2);
  unsigned short* refwb = (unsigned short*)alloc((size_t)256 * 640 * 2);
  float* accum          = (float*)alloc(8 * sizeof(float));
  // activations
  float* x0    = (float*)alloc((size_t)BB * 3 * HH * WW * 4);
  unsigned short* col = (unsigned short*)alloc((size_t)M2 * K2 * 2);  // largest im2col
  float* c1    = (float*)alloc((size_t)M1 * N1 * 4);
  float* c2    = (float*)alloc((size_t)M2 * N2 * 4);
  float* f0pre = (float*)alloc((size_t)SEQE * 4);
  float* f1pre = (float*)alloc((size_t)SEQE * 4);
  float* f0    = (float*)alloc((size_t)SEQE * 4);
  float* f1    = (float*)alloc((size_t)SEQE * 4);
  unsigned short* f0b = (unsigned short*)alloc((size_t)SEQE * 2);
  unsigned short* f1b = (unsigned short*)alloc((size_t)SEQE * 2);
  EncTmp t;
  t.q    = (float*)alloc((size_t)SEQE * 4);
  t.k    = (float*)alloc((size_t)SEQE * 4);
  t.v    = (float*)alloc((size_t)SEQE * 4);
  t.msg  = (float*)alloc((size_t)SEQE * 4);
  t.ffn1 = (float*)alloc((size_t)BB * LSEQ * 640 * 4);
  t.ffn2 = (float*)alloc((size_t)SEQE * 4);
  t.msgb  = (unsigned short*)alloc((size_t)SEQE * 2);
  t.cat1  = (unsigned short*)alloc((size_t)BB * LSEQ * 640 * 2);
  t.ffn1b = (unsigned short*)alloc((size_t)BB * LSEQ * 640 * 2);
  t.kvbuf   = (float*)alloc((size_t)BB * NHD * DHD * DHD * 4);
  t.ksumbuf = (float*)alloc((size_t)BB * NHD * DHD * 4);
  float* sim            = (float*)alloc((size_t)SIME * 4);
  unsigned short* probs = (unsigned short*)alloc((size_t)SIME * 2);
  float* warped         = (float*)alloc((size_t)SEQE * 4);
  unsigned short* cat2  = (unsigned short*)alloc((size_t)BB * LSEQ * 640 * 2);
  float* oseq           = (float*)alloc((size_t)BB * LSEQ * 192 * 4);
  float* rseq           = (float*)alloc((size_t)BB * LSEQ * 256 * 4);

  // --- weight conversions (re-done every call: deterministic) ---
  launch_cvt(stream, bb_w1, bw1, (long long)N1 * K1);
  launch_cvt(stream, bb_w2, bw2, (long long)N2 * K2);
  launch_cvt(stream, bb_w3, bw3, (long long)N3 * K3);
  launch_cvt(stream, Wq, wqb, 4LL * DD * DD);
  launch_cvt(stream, Wk, wkb, 4LL * DD * DD);
  launch_cvt(stream, Wv, wvb, 4LL * DD * DD);
  launch_cvt(stream, Wm, wmb, 4LL * DD * DD);
  launch_cvt(stream, W1, w1b, 4LL * 640 * 640);
  launch_cvt(stream, W2, w2b, 4LL * 640 * 320);
  launch_cvt(stream, out_w, outwb, 192LL * 320);
  launch_cvt(stream, ref_w, refwb, 256LL * 640);
  k_zero<<<1, 32, 0, stream>>>(accum, 8);

  // --- masked input ---
  k_mask_fill<<<gdiv((long long)BB * 3 * HH * WW, 256), 256, 0, stream>>>(
      img, mask, mtoken, x0, (long long)BB * 3 * HH * WW);

  // --- backbone: im2col + WMMA GEMM (weights OIHW == [N,K] -> transB) ---
  k_im2col<<<gdiv((long long)M1 * K1, 256), 256, 0, stream>>>(x0, col, BB, 3, HH, WW, 256, 256, 1);
  launch_gemm(stream, col, bw1, c1, nullptr, bb_b1, M1, N1, K1, 1, 1, 0, 0, 0, 1);
  k_im2col<<<gdiv((long long)M2 * K2, 256), 256, 0, stream>>>(c1, col, BB, 128, 256, 256, 128, 128, 0);
  launch_gemm(stream, col, bw2, c2, nullptr, bb_b2, M2, N2, K2, 1, 1, 0, 0, 0, 1);
  k_im2col<<<gdiv((long long)M3 * K3, 256), 256, 0, stream>>>(c2, col, BB, 256, 128, 128, 64, 64, 0);
  launch_gemm(stream, col, bw3, f0pre, nullptr, bb_b3, M3, N3, K3, 1, 1, 0, 0, 0, 1);
  k_posenc<<<gdiv(SEQE, 256), 256, 0, stream>>>(f0pre, f0, f0b);

  k_im2col<<<gdiv((long long)M1 * K1, 256), 256, 0, stream>>>(img2, col, BB, 3, HH, WW, 256, 256, 1);
  launch_gemm(stream, col, bw1, c1, nullptr, bb_b1, M1, N1, K1, 1, 1, 0, 0, 0, 1);
  k_im2col<<<gdiv((long long)M2 * K2, 256), 256, 0, stream>>>(c1, col, BB, 128, 256, 256, 128, 128, 0);
  launch_gemm(stream, col, bw2, c2, nullptr, bb_b2, M2, N2, K2, 1, 1, 0, 0, 0, 1);
  k_im2col<<<gdiv((long long)M3 * K3, 256), 256, 0, stream>>>(c2, col, BB, 256, 128, 128, 64, 64, 0);
  launch_gemm(stream, col, bw3, f1pre, nullptr, bb_b3, M3, N3, K3, 1, 1, 0, 0, 0, 1);
  k_posenc<<<gdiv(SEQE, 256), 256, 0, stream>>>(f1pre, f1, f1b);

  // --- transformer: self, cross, self, cross ---
  for (int i = 0; i < 4; ++i) {
    const unsigned short* wq = wqb + (size_t)i * DD * DD;
    const unsigned short* wk = wkb + (size_t)i * DD * DD;
    const unsigned short* wv = wvb + (size_t)i * DD * DD;
    const unsigned short* wm = wmb + (size_t)i * DD * DD;
    const unsigned short* w1 = w1b + (size_t)i * 640 * 640;
    const unsigned short* w2 = w2b + (size_t)i * 640 * 320;
    const float* g1i = g1 + i * DD; const float* b1i = b1 + i * DD;
    const float* g2i = g2 + i * DD; const float* b2i = b2 + i * DD;
    if ((i & 1) == 0) {  // self
      run_encoder(stream, f0, f0b, f0b, wq, wk, wv, wm, w1, w2, g1i, b1i, g2i, b2i, t);
      run_encoder(stream, f1, f1b, f1b, wq, wk, wv, wm, w1, w2, g1i, b1i, g2i, b2i, t);
    } else {             // cross (f1 uses the just-updated f0, as in the reference)
      run_encoder(stream, f0, f0b, f1b, wq, wk, wv, wm, w1, w2, g1i, b1i, g2i, b2i, t);
      run_encoder(stream, f1, f1b, f0b, wq, wk, wv, wm, w1, w2, g1i, b1i, g2i, b2i, t);
    }
  }

  // --- similarity / softmax / warp (batched WMMA GEMMs) ---
  launch_gemm(stream, f0b, f1b, sim, nullptr, nullptr, LSEQ, LSEQ, DD, 1, 0,
              (long long)LSEQ * DD, (long long)LSEQ * DD, (long long)LSEQ * LSEQ, BB);
  k_softmax_bf16<<<BB * LSEQ, 256, 0, stream>>>(sim, probs, LSEQ, 10.f /* 1/TEMP */);
  launch_gemm(stream, probs, f1b, warped, nullptr, nullptr, LSEQ, DD, LSEQ, 0, 0,
              (long long)LSEQ * LSEQ, (long long)LSEQ * DD, (long long)LSEQ * DD, BB);

  // --- heads (1x1 convs == seq-layout GEMMs with transB weights) ---
  k_concat_bf16<<<gdiv((long long)BB * LSEQ * 640, 256), 256, 0, stream>>>(
      f0, warped, cat2, (long long)BB * LSEQ, DD, DD);
  launch_gemm(stream, f0b,  outwb, oseq, nullptr, out_b, BB * LSEQ, 192, 320, 1, 0, 0, 0, 0, 1);
  launch_gemm(stream, cat2, refwb, rseq, nullptr, ref_b, BB * LSEQ, 256, 640, 1, 0, 0, 0, 0, 1);

  // --- pixel shuffle, refinement, losses ---
  float* rgb     = out + 1;
  float* refined = out + 1 + (long long)BB * 3 * HH * WW;
  float* rel     = out + 1 + 2LL * BB * 3 * HH * WW;
  k_pixshuf_rgb<<<gdiv((long long)BB * 3 * HH * WW, 256), 256, 0, stream>>>(oseq, rgb);
  k_refine<<<gdiv((long long)BB * HH * WW, 256), 256, 0, stream>>>(rseq, rgb, refined, rel);
  k_loss_reduce<<<gdiv((long long)BB * HH * WW, 256), 256, 0, stream>>>(
      img, rgb, refined, mask, masksup, accum);
  k_finalize<<<1, 32, 0, stream>>>(accum, out);
}
